// _ConvertedUnidirectionalBandRNN_19490561589975
// MI455X (gfx1250) — matrix operand order repaired and checked by reading, not compiled, not run on hardware
//
#include <hip/hip_runtime.h>
#include <math.h>

typedef __bf16 bf16x16 __attribute__((ext_vector_type(16)));
typedef float  f32x8   __attribute__((ext_vector_type(8)));
typedef float  f32x4   __attribute__((ext_vector_type(4)));

namespace {
constexpr int kBands     = 32;
constexpr int kDin       = 128;
constexpr int kH         = 128;
constexpr int kFeat      = kBands * kDin;   // 4096
constexpr int kRows      = 16 * 512;        // 8192 (= B*T)
constexpr int kCTiles    = 24;              // 384 gate cols / 16
constexpr int kKSteps    = 4;               // 128 / 32 (bf16 WMMA K)
constexpr int kRowsWG    = 32;              // rows per workgroup
constexpr int kThreads   = 256;             // 8 waves (wave32)
constexpr int kFragElems = 512;             // bf16 per B-fragment (32 lanes x 16)
}

// ---------------------------------------------------------------------------
// Pack weight_ih / weight_hh (f32, [384,128] row-major, gi = x @ W^T) into
// bf16 WMMA B-fragments.  B is KxN with element (k, n) = W[n, k].
// Dense 16x16x32 bf16 B layout (wave32): lane half h holds K=8h..8h+7 (elems 0-7)
// and K=16+8h..16+8h+7 (elems 8-15) for N=lane%16.
// Fragment (wsel, c, s) lives at ws + ((wsel*24 + c)*4 + s)*512 bf16.
// ---------------------------------------------------------------------------
__global__ void pack_weights_kernel(const float* __restrict__ w_ih,
                                    const float* __restrict__ w_hh,
                                    __bf16* __restrict__ wsbf) {
  const int f    = blockIdx.x;                 // 0 .. 191
  const int lane = threadIdx.x;                // 0 .. 31
  const int wsel = f / (kCTiles * kKSteps);
  const int rem  = f % (kCTiles * kKSteps);
  const int c    = rem / kKSteps;
  const int s    = rem % kKSteps;
  const float* W = wsel ? w_hh : w_ih;

  const int ncol = c * 16 + (lane & 15);       // gate (output) column
  const int kb   = 32 * s + (lane >> 4) * 8;   // K base for this lane half

  __bf16* dst = wsbf + (size_t)f * kFragElems + lane * 16;
#pragma unroll
  for (int j = 0; j < 16; ++j)
    dst[j] = (__bf16)W[ncol * kDin + kb + (j < 8 ? j : j + 8)];
}

__device__ __forceinline__ bf16x16 load_bfrag(const __bf16* base, int c, int s,
                                              int lane) {
  return *(const bf16x16*)(base + ((size_t)c * kKSteps + s) * kFragElems +
                           (size_t)lane * 16);
}

// ---------------------------------------------------------------------------
// Fused band-GRU: workgroup = 32 rows, 8 waves; wave wv handles row-group
// (wv&1) and gate-column tiles j = (wv>>1)*2 .. +1 of each of r/z/n.
// Hidden state is double-buffered in LDS (f32); A operands converted to bf16
// at fragment-gather time.  2 barriers per band.
// ---------------------------------------------------------------------------
__launch_bounds__(kThreads)
__global__ void band_gru_kernel(const float* __restrict__ x,
                                const __bf16* __restrict__ wfrag,
                                const float* __restrict__ bias_ih,
                                const float* __restrict__ bias_hh,
                                float* __restrict__ y) {
  __shared__ float ldsH[2][kRowsWG * kH];  // 2 x 16 KB hidden state (cur/nxt)
  __shared__ float ldsX[kRowsWG * kDin];   // 16 KB staged x block

  const int tid  = threadIdx.x;
  const int lane = tid & 31;
  const int wv   = tid >> 5;        // 0..7
  const int rg   = wv & 1;          // row-group (16 rows) within WG
  const int j0   = (wv >> 1) * 2;   // first of two owned column tiles
  const int mcol = lane & 15;       // N column / M row selector
  const int hi   = lane >> 4;       // lane half

  const int rowBase = blockIdx.x * kRowsWG;

  // cooperative loader geometry (32 rows x 128 cols, 16 f32 per thread)
  const int ldRow = tid >> 3;        // 0..31
  const int ldSeg = (tid & 7) * 16;  // 0..112

  for (int i = tid; i < kRowsWG * kH; i += kThreads) ldsH[0][i] = 0.0f;
  __syncthreads();

  const __bf16* fih = wfrag;
  const __bf16* fhh = wfrag + (size_t)kCTiles * kKSteps * kFragElems;

#pragma unroll 1
  for (int b = 0; b < kBands; ++b) {
    const float* hcur = ldsH[b & 1];
    float*       hnxt = ldsH[(b & 1) ^ 1];

    // ---- stage x block for this band into LDS (coalesced b128 traffic) ----
    {
      const float* src = x + (size_t)(rowBase + ldRow) * kFeat + b * kDin + ldSeg;
      float* dst = ldsX + ldRow * kDin + ldSeg;
#pragma unroll
      for (int q = 0; q < 4; ++q)
        *(f32x4*)(dst + 4 * q) = *(const f32x4*)(src + 4 * q);
      if (b + 1 < kBands) __builtin_prefetch(src + kDin, 0, 1);
    }
    __syncthreads();

    // ---- gather A fragments (16x32 bf16 layout) for x and h ----
    // lane half hi: VGPRs 0-3 hold K = d0..d0+7, VGPRs 4-7 hold K = d0+16..+23
    bf16x16 ax[kKSteps], ah[kKSteps];
    {
      const float* xr = ldsX + (rg * 16 + mcol) * kDin;
      const float* hr = hcur + (rg * 16 + mcol) * kH;
#pragma unroll
      for (int s = 0; s < kKSteps; ++s) {
        const int d0 = 32 * s + hi * 8;
        const f32x4 a0 = *(const f32x4*)(xr + d0);
        const f32x4 a1 = *(const f32x4*)(xr + d0 + 4);
        const f32x4 a2 = *(const f32x4*)(xr + d0 + 16);
        const f32x4 a3 = *(const f32x4*)(xr + d0 + 20);
        const f32x4 h0 = *(const f32x4*)(hr + d0);
        const f32x4 h1 = *(const f32x4*)(hr + d0 + 4);
        const f32x4 h2 = *(const f32x4*)(hr + d0 + 16);
        const f32x4 h3 = *(const f32x4*)(hr + d0 + 20);
#pragma unroll
        for (int e = 0; e < 4; ++e) {
          ax[s][e]      = (__bf16)a0[e];
          ax[s][4 + e]  = (__bf16)a1[e];
          ax[s][8 + e]  = (__bf16)a2[e];
          ax[s][12 + e] = (__bf16)a3[e];
          ah[s][e]      = (__bf16)h0[e];
          ah[s][4 + e]  = (__bf16)h1[e];
          ah[s][8 + e]  = (__bf16)h2[e];
          ah[s][12 + e] = (__bf16)h3[e];
        }
      }
    }

    // ---- GEMMs + gates, one owned column tile at a time ----
#pragma unroll 1
    for (int t = 0; t < 2; ++t) {
      const int j = j0 + t;
      const float br  = bias_ih[j * 16 + mcol]       + bias_hh[j * 16 + mcol];
      const float bz  = bias_ih[(8 + j) * 16 + mcol] + bias_hh[(8 + j) * 16 + mcol];
      const float bin = bias_ih[(16 + j) * 16 + mcol];
      const float bhn = bias_hh[(16 + j) * 16 + mcol];

      f32x8 accr, accz, gin, ghn;
#pragma unroll
      for (int e = 0; e < 8; ++e) { accr[e] = br; accz[e] = bz; gin[e] = bin; ghn[e] = bhn; }

      // r gate: x-side then h-side (per-gate K loops bound live B-fragments)
#pragma unroll
      for (int s = 0; s < kKSteps; ++s)
        accr = __builtin_amdgcn_wmma_f32_16x16x32_bf16(
            false, ax[s], false, load_bfrag(fih, j, s, lane), (short)0, accr, false, false);
#pragma unroll
      for (int s = 0; s < kKSteps; ++s)
        accr = __builtin_amdgcn_wmma_f32_16x16x32_bf16(
            false, ah[s], false, load_bfrag(fhh, j, s, lane), (short)0, accr, false, false);
      // z gate
#pragma unroll
      for (int s = 0; s < kKSteps; ++s)
        accz = __builtin_amdgcn_wmma_f32_16x16x32_bf16(
            false, ax[s], false, load_bfrag(fih, 8 + j, s, lane), (short)0, accz, false, false);
#pragma unroll
      for (int s = 0; s < kKSteps; ++s)
        accz = __builtin_amdgcn_wmma_f32_16x16x32_bf16(
            false, ah[s], false, load_bfrag(fhh, 8 + j, s, lane), (short)0, accz, false, false);
      // n gate: input and hidden sides must stay separate
#pragma unroll
      for (int s = 0; s < kKSteps; ++s)
        gin = __builtin_amdgcn_wmma_f32_16x16x32_bf16(
            false, ax[s], false, load_bfrag(fih, 16 + j, s, lane), (short)0, gin, false, false);
#pragma unroll
      for (int s = 0; s < kKSteps; ++s)
        ghn = __builtin_amdgcn_wmma_f32_16x16x32_bf16(
            false, ah[s], false, load_bfrag(fhh, 16 + j, s, lane), (short)0, ghn, false, false);

      // previous h for these (M,N) elements (C layout: M = r + 8*hi, N = mcol)
      const int colIdx = j * 16 + mcol;
      const int rowOff = rg * 16 + hi * 8;
#pragma unroll
      for (int r = 0; r < 8; ++r) {
        const float hold = hcur[(rowOff + r) * kH + colIdx];
        const float rr = 1.0f / (1.0f + __expf(-accr[r]));
        const float zz = 1.0f / (1.0f + __expf(-accz[r]));
        const float nn = tanhf(gin[r] + rr * ghn[r]);
        hnxt[(rowOff + r) * kH + colIdx] = (1.0f - zz) * nn + zz * hold;
      }
    }
    __syncthreads();  // all reads of cur / ldsX done; all writes to nxt done

    // ---- write y for this band (coalesced from LDS) ----
    {
      float* dst = y + (size_t)(rowBase + ldRow) * kFeat + b * kDin + ldSeg;
      const float* srcH = hnxt + ldRow * kH + ldSeg;
#pragma unroll
      for (int q = 0; q < 4; ++q)
        *(f32x4*)(dst + 4 * q) = *(const f32x4*)(srcH + 4 * q);
    }
    // next band's h writes go to the other buffer and happen only after the
    // top-of-band barrier, so no extra barrier is needed here.
  }
}

// ---------------------------------------------------------------------------
extern "C" void kernel_launch(void* const* d_in, const int* in_sizes, int n_in,
                              void* d_out, int out_size, void* d_ws, size_t ws_size,
                              hipStream_t stream) {
  const float* x    = (const float*)d_in[0];
  const float* w_ih = (const float*)d_in[1];
  const float* w_hh = (const float*)d_in[2];
  const float* b_ih = (const float*)d_in[3];
  const float* b_hh = (const float*)d_in[4];
  float* y = (float*)d_out;
  __bf16* wfrag = (__bf16*)d_ws;  // 2 * 24 * 4 * 512 bf16 = 192 KB

  pack_weights_kernel<<<2 * kCTiles * kKSteps, 32, 0, stream>>>(w_ih, w_hh, wfrag);
  band_gru_kernel<<<kRows / kRowsWG, kThreads, 0, stream>>>(x, wfrag, b_ih, b_hh, y);
}
